// MultiHeadCrossAttention_89086211654111
// MI455X (gfx1250) — matrix operand order repaired and checked by reading, not compiled, not run on hardware
//
#include <hip/hip_runtime.h>
#include <hip/hip_bf16.h>

// ---------------------------------------------------------------------------
// CDNA5 (gfx1250) multi-head cross attention, bf16 WMMA pipeline.
//   pre:  x, xa, Wq, Wk, Wv, Wo  ->  bf16 copies (one elementwise pass)
//   q = (x @ Wq^T + bq) * s        (bf16, ws)
//   k = (xa @ Wk^T) * s            (bf16, ws)
//   v =  xa @ Wv^T + bv            (bf16, ws)
//   attn = softmax(q k^T) v        (flash-style, bf16, ws)
//   out  = attn @ Wo^T + bo        (f32, d_out)
//
// Every 8-VGPR WMMA operand is exactly two 16-byte (b128) loads:
//   A-layout: VGPR0-7 = K[16h .. 16h+15]   (same k order as B)
//   B-layout: VGPR0-7 = K[16h .. 16h+15]            (h = lane>>4)
// GEMM tiles are staged with double-buffered global_load_async_to_lds_b128
// (ASYNCcnt); attention K chunks use the same async path. Softmax row-sums
// are computed with an extra WMMA against an all-ones B fragment. Each
// attention wave owns 32 queries so every K/V fragment feeds two WMMAs.
// ---------------------------------------------------------------------------

typedef __attribute__((ext_vector_type(16))) __bf16 v16bf;
typedef __attribute__((ext_vector_type(8)))  float  v8f;

union Frag16 {
    v16bf v;
    uint4 q[2];
    unsigned int d[8];
};

static __device__ inline v8f wmma_bf16(const Frag16& a, const Frag16& b, v8f c) {
    return __builtin_amdgcn_wmma_f32_16x16x32_bf16(
        /*neg_a=*/false, a.v, /*neg_b=*/false, b.v,
        /*c_mod=*/(short)0, c, /*reuse_a=*/false, /*reuse_b=*/false);
}

static __device__ inline unsigned short f2bf(float f) {
    unsigned u = __float_as_uint(f);
    unsigned r = u + 0x7FFFu + ((u >> 16) & 1u);   // round-to-nearest-even
    return (unsigned short)(r >> 16);
}

// ---------------------------------------------------------------------------
// f32 -> bf16 pack, 8 elements/thread (n is always a multiple of 2048 here).
// ---------------------------------------------------------------------------
__global__ __launch_bounds__(256) void cvt_bf16_kernel(
    const float* __restrict__ in, unsigned short* __restrict__ out, int n)
{
    int i = (blockIdx.x * 256 + threadIdx.x) * 8;
    if (i >= n) return;
    float4 a = *(const float4*)(in + i);
    float4 b = *(const float4*)(in + i + 4);
    uint4 r;
    r.x = (unsigned)f2bf(a.x) | ((unsigned)f2bf(a.y) << 16);
    r.y = (unsigned)f2bf(a.z) | ((unsigned)f2bf(a.w) << 16);
    r.z = (unsigned)f2bf(b.x) | ((unsigned)f2bf(b.y) << 16);
    r.w = (unsigned)f2bf(b.z) | ((unsigned)f2bf(b.w) << 16);
    *(uint4*)(out + i) = r;
}

// ---------------------------------------------------------------------------
// C[M,N] = scale * (A[M,K] @ W[N,K]^T + bias[N]), A and W in bf16.
// Block: 256 threads (8 waves), 64x64 tile, K-step 32.
// Double-buffered async LDS staging: tile i+1 copies overlap tile i WMMAs.
// LDS row stride 40 u16 = 80B: 16B-aligned rows, 5 quad-banks, conflict-free.
// ---------------------------------------------------------------------------
template <bool OUT_IS_BF16>
__global__ __launch_bounds__(256) void gemm_bt(
    const unsigned short* __restrict__ A, const unsigned short* __restrict__ Wb,
    const float* __restrict__ bias, float scale,
    void* __restrict__ Cv, int M, int Nn, int K)
{
    __shared__ alignas(16) unsigned short As[2][64][40];
    __shared__ alignas(16) unsigned short Bs[2][64][40];

    const int tid  = threadIdx.x;
    const int lane = tid & 31;
    const int wave = tid >> 5;
    const int lm   = lane & 15;
    const int half = lane >> 4;
    const int mrow = wave & 3;     // 16-row strip of the 64-row tile
    const int nhalf = wave >> 2;   // 32-col half of the 64-col tile
    const int m0 = blockIdx.x * 64;
    const int n0 = blockIdx.y * 64;

    const int lr = tid >> 2;          // staging row 0..63
    const int lc = (tid & 3) * 8;     // staging col base 0/8/16/24

    int gm = m0 + lr; if (gm >= M) gm = M - 1;
    const unsigned short* aSrc = A  + (size_t)gm * K + lc;
    const unsigned short* wSrc = Wb + (size_t)(n0 + lr) * K + lc;

    auto issue = [&](int kt, int buf) {
        unsigned la = (unsigned)(unsigned long long)&As[buf][lr][lc];
        unsigned lb = (unsigned)(unsigned long long)&Bs[buf][lr][lc];
        const unsigned short* pa = aSrc + kt;
        const unsigned short* pw = wSrc + kt;
        asm volatile("global_load_async_to_lds_b128 %0, %1, off"
                     :: "v"(la), "v"(pa) : "memory");
        asm volatile("global_load_async_to_lds_b128 %0, %1, off"
                     :: "v"(lb), "v"(pw) : "memory");
    };

    v8f acc[2];
    acc[0] = v8f{}; acc[1] = v8f{};

    issue(0, 0);
    const int steps = K / 32;
    for (int it = 0; it < steps; ++it) {
        const int buf = it & 1;
        __syncthreads();                       // readers of buf^1 are done
        if (it + 1 < steps) {
            issue((it + 1) * 32, buf ^ 1);     // overlap with this tile's math
            asm volatile("s_wait_asynccnt 0x2" ::: "memory");  // tile `it` landed
        } else {
            asm volatile("s_wait_asynccnt 0x0" ::: "memory");
        }
        __syncthreads();                       // all waves see buf complete

        Frag16 a;
        a.q[0] = *(const uint4*)&As[buf][mrow * 16 + lm][16 * half];
        a.q[1] = *(const uint4*)&As[buf][mrow * 16 + lm][16 * half + 8];
        Frag16 b0, b1;
        b0.q[0] = *(const uint4*)&Bs[buf][(nhalf * 2 + 0) * 16 + lm][half * 16];
        b0.q[1] = *(const uint4*)&Bs[buf][(nhalf * 2 + 0) * 16 + lm][half * 16 + 8];
        b1.q[0] = *(const uint4*)&Bs[buf][(nhalf * 2 + 1) * 16 + lm][half * 16];
        b1.q[1] = *(const uint4*)&Bs[buf][(nhalf * 2 + 1) * 16 + lm][half * 16 + 8];

        acc[0] = wmma_bf16(a, b0, acc[0]);
        acc[1] = wmma_bf16(a, b1, acc[1]);
    }

    // ---- epilogue: C layout row = g + half*8, col = lm ----
    #pragma unroll
    for (int nc = 0; nc < 2; ++nc) {
        #pragma unroll
        for (int g = 0; g < 8; ++g) {
            int m = m0 + mrow * 16 + half * 8 + g;
            int n = n0 + (nhalf * 2 + nc) * 16 + lm;
            if (m < M) {
                float val = acc[nc][g];
                if (bias) val += bias[n];
                val *= scale;
                if constexpr (OUT_IS_BF16)
                    ((unsigned short*)Cv)[(size_t)m * Nn + n] = f2bf(val);
                else
                    ((float*)Cv)[(size_t)m * Nn + n] = val;
            }
        }
    }
}

// ---------------------------------------------------------------------------
// Flash attention: block = 8 waves sharing one (batch, head), 256 queries.
// Each wave owns TWO 16-row Q tiles, so every K/V B-fragment in registers
// feeds two WMMAs. K chunk staged via global_load_async_to_lds_b128; V chunk
// staged transposed [d][s]; P row-sums via WMMA against an all-ones fragment.
// ---------------------------------------------------------------------------
__global__ __launch_bounds__(256) void attn_kernel(
    const unsigned short* __restrict__ Q,    // bf16 (4,1024,1024), pre-scaled
    const unsigned short* __restrict__ Km,   // bf16 (4,1500,1024), pre-scaled
    const unsigned short* __restrict__ Vm,   // bf16 (4,1500,1024)
    unsigned short* __restrict__ Ob,         // bf16 (4,1024,1024)
    int T, int T1, int D, int HD)
{
    __shared__ alignas(16) unsigned short Kt[32][72];      // [s][d], 144B rows
    __shared__ alignas(16) unsigned short Vt[64][40];      // [d][s], 80B rows
    __shared__ alignas(16) unsigned short Pb[8][2][16][40];

    const int tid  = threadIdx.x;
    const int lane = tid & 31;
    const int wave = tid >> 5;
    const int lm   = lane & 15;
    const int half = lane >> 4;

    const int blk = blockIdx.x;
    const int qb  = blk & 3;             // T/256 = 4 query blocks
    const int h   = (blk >> 2) & 15;
    const int n   = blk >> 6;
    const int t0  = qb * 256 + wave * 32;

    // ---- Q fragments: 2 tiles x (d 0..31, 32..63) ----
    Frag16 qa[2][2];
    #pragma unroll
    for (int qt = 0; qt < 2; ++qt) {
        const unsigned short* qp =
            Q + ((size_t)(n * T + t0 + qt * 16 + lm)) * D + h * HD;
        qa[qt][0].q[0] = *(const uint4*)(qp + 16 * half);
        qa[qt][0].q[1] = *(const uint4*)(qp + 16 * half + 8);
        qa[qt][1].q[0] = *(const uint4*)(qp + 32 + 16 * half);
        qa[qt][1].q[1] = *(const uint4*)(qp + 32 + 16 * half + 8);
    }

    Frag16 ones;                                   // bf16 1.0 pairs
    #pragma unroll
    for (int i = 0; i < 8; ++i) ones.d[i] = 0x3F803F80u;

    float mrow[2][8], lrow[2][8];
    #pragma unroll
    for (int qt = 0; qt < 2; ++qt)
        #pragma unroll
        for (int g = 0; g < 8; ++g) { mrow[qt][g] = -3.0e38f; lrow[qt][g] = 0.0f; }
    v8f oacc[2][4];
    #pragma unroll
    for (int qt = 0; qt < 2; ++qt)
        #pragma unroll
        for (int d4 = 0; d4 < 4; ++d4) oacc[qt][d4] = v8f{};

    const int srow = tid >> 3;         // staging row 0..31
    const int scol = (tid & 7) * 8;    // staging col base (16B chunks)

    const int nch = (T1 + 31) / 32;
    for (int ch = 0; ch < nch; ++ch) {
        __syncthreads();               // previous chunk's LDS reads done
        {
            int s  = ch * 32 + srow;
            int sg = (s < T1) ? s : (T1 - 1);
            // K: async byte-copy global -> LDS (b128 per lane, ASYNCcnt)
            {
                const unsigned short* kp =
                    Km + ((size_t)(n * T1 + sg)) * D + h * HD + scol;
                unsigned ldsoff = (unsigned)(unsigned long long)&Kt[srow][scol];
                asm volatile("global_load_async_to_lds_b128 %0, %1, off"
                             :: "v"(ldsoff), "v"(kp) : "memory");
            }
            // V: b128 global load, transposed scatter into Vt[d][s]
            {
                uint4 vraw = *(const uint4*)(Vm + ((size_t)(n * T1 + sg)) * D +
                                             h * HD + scol);
                const unsigned short* ve = (const unsigned short*)&vraw;
                #pragma unroll
                for (int j = 0; j < 8; ++j) Vt[scol + j][srow] = ve[j];
            }
            asm volatile("s_wait_asynccnt 0x0" ::: "memory");
        }
        __syncthreads();

        // ---- scores: S = Q(32x64) @ K_chunk^T(64x32); K fragments shared ----
        v8f sacc[2][2];
        #pragma unroll
        for (int sc = 0; sc < 2; ++sc) {
            const int sl = sc * 16 + lm;
            Frag16 b0, b1;
            b0.q[0] = *(const uint4*)&Kt[sl][half * 16];
            b0.q[1] = *(const uint4*)&Kt[sl][half * 16 + 8];
            b1.q[0] = *(const uint4*)&Kt[sl][32 + half * 16];
            b1.q[1] = *(const uint4*)&Kt[sl][32 + half * 16 + 8];
            #pragma unroll
            for (int qt = 0; qt < 2; ++qt) {
                v8f s = v8f{};
                s = wmma_bf16(qa[qt][0], b0, s);
                s = wmma_bf16(qa[qt][1], b1, s);
                sacc[qt][sc] = s;
            }
        }

        // ---- tail mask: uniform branch, only the last chunk pays it ----
        if ((ch + 1) * 32 > T1) {
            bool msk0 = (ch * 32 + lm) >= T1;
            bool msk1 = (ch * 32 + 16 + lm) >= T1;
            #pragma unroll
            for (int qt = 0; qt < 2; ++qt)
                #pragma unroll
                for (int g = 0; g < 8; ++g) {
                    if (msk0) sacc[qt][0][g] = -3.0e38f;
                    if (msk1) sacc[qt][1][g] = -3.0e38f;
                }
        }

        // ---- online softmax per q-tile: butterfly max, exp, store P ----
        float alpha8[2][8];
        #pragma unroll
        for (int qt = 0; qt < 2; ++qt) {
            #pragma unroll
            for (int g = 0; g < 8; ++g) {
                float mx = fmaxf(sacc[qt][0][g], sacc[qt][1][g]);
                #pragma unroll
                for (int off = 8; off >= 1; off >>= 1)
                    mx = fmaxf(mx, __shfl_xor(mx, off, 32));
                float mnew = fmaxf(mrow[qt][g], mx);
                alpha8[qt][g] = __expf(mrow[qt][g] - mnew);
                mrow[qt][g] = mnew;
                Pb[wave][qt][half * 8 + g][lm] =
                    f2bf(__expf(sacc[qt][0][g] - mnew));
                Pb[wave][qt][half * 8 + g][16 + lm] =
                    f2bf(__expf(sacc[qt][1][g] - mnew));
            }
        }

        // ---- reload P tiles in A-layout + row sums via WMMA ----
        Frag16 pa[2];
        #pragma unroll
        for (int qt = 0; qt < 2; ++qt) {
            pa[qt].q[0] = *(const uint4*)&Pb[wave][qt][lm][16 * half];
            pa[qt].q[1] = *(const uint4*)&Pb[wave][qt][lm][16 * half + 8];
            v8f rsum = wmma_bf16(pa[qt], ones, v8f{});
            #pragma unroll
            for (int g = 0; g < 8; ++g) {
                lrow[qt][g] = lrow[qt][g] * alpha8[qt][g] + rsum[g];
                #pragma unroll
                for (int d4 = 0; d4 < 4; ++d4) oacc[qt][d4][g] *= alpha8[qt][g];
            }
        }

        // ---- O += P(32x32) @ V_chunk(32x64); V fragments shared ----
        #pragma unroll
        for (int d4 = 0; d4 < 4; ++d4) {
            Frag16 vb;
            vb.q[0] = *(const uint4*)&Vt[d4 * 16 + lm][half * 16];
            vb.q[1] = *(const uint4*)&Vt[d4 * 16 + lm][half * 16 + 8];
            #pragma unroll
            for (int qt = 0; qt < 2; ++qt)
                oacc[qt][d4] = wmma_bf16(pa[qt], vb, oacc[qt][d4]);
        }
    }

    // ---- normalize and store (N, T, H*d) as bf16 ----
    #pragma unroll
    for (int qt = 0; qt < 2; ++qt) {
        #pragma unroll
        for (int g = 0; g < 8; ++g) {
            float inv = (lrow[qt][g] > 0.0f) ? (1.0f / lrow[qt][g]) : 0.0f;
            int t = t0 + qt * 16 + half * 8 + g;
            size_t base = ((size_t)(n * T + t)) * D + h * HD;
            #pragma unroll
            for (int d4 = 0; d4 < 4; ++d4)
                Ob[base + d4 * 16 + lm] = f2bf(oacc[qt][d4][g] * inv);
        }
    }
}

// ---------------------------------------------------------------------------
extern "C" void kernel_launch(void* const* d_in, const int* in_sizes, int n_in,
                              void* d_out, int out_size, void* d_ws, size_t ws_size,
                              hipStream_t stream) {
    const float* x  = (const float*)d_in[0];   // (4,1024,1024)
    const float* xa = (const float*)d_in[1];   // (4,1500,1024)
    const float* Wq = (const float*)d_in[2];
    const float* bq = (const float*)d_in[3];
    const float* Wk = (const float*)d_in[4];
    const float* Wv = (const float*)d_in[5];
    const float* bv = (const float*)d_in[6];
    const float* Wo = (const float*)d_in[7];
    const float* bo = (const float*)d_in[8];

    const int D = 1024, T = 1024, T1 = 1500, NB = 4, HD = 64;
    const int Mq = NB * T;    // 4096
    const int Mk = NB * T1;   // 6000
    const float scale = 0.35355339059327373f;  // 64^-0.25

    // ---- workspace layout (bytes), all 16B aligned ----
    unsigned char* ws = (unsigned char*)d_ws;
    const size_t qB = (size_t)Mq * D * 2;     // 8,388,608
    const size_t kB = (size_t)Mk * D * 2;     // 12,288,000
    const size_t wB = (size_t)D * D * 2;      // 2,097,152
    unsigned short* qv  = (unsigned short*)(ws);
    unsigned short* kv  = (unsigned short*)(ws + qB);
    unsigned short* vv  = (unsigned short*)(ws + qB + kB);
    unsigned short* av  = (unsigned short*)(ws + qB + 2 * kB);
    unsigned short* xb  = (unsigned short*)(ws + 2 * qB + 2 * kB);
    unsigned short* xab = (unsigned short*)(ws + 3 * qB + 2 * kB);
    unsigned short* wqb = (unsigned short*)(ws + 3 * qB + 3 * kB);
    unsigned short* wkb = (unsigned short*)(ws + 3 * qB + 3 * kB + wB);
    unsigned short* wvb = (unsigned short*)(ws + 3 * qB + 3 * kB + 2 * wB);
    unsigned short* wob = (unsigned short*)(ws + 3 * qB + 3 * kB + 3 * wB);

    dim3 blk(256);

    // ---- one-shot f32 -> bf16 conversions ----
    const int nX = Mq * D, nXA = Mk * D, nW = D * D;
    cvt_bf16_kernel<<<dim3(nX  / 2048), blk, 0, stream>>>(x,  xb,  nX);
    cvt_bf16_kernel<<<dim3(nXA / 2048), blk, 0, stream>>>(xa, xab, nXA);
    cvt_bf16_kernel<<<dim3(nW  / 2048), blk, 0, stream>>>(Wq, wqb, nW);
    cvt_bf16_kernel<<<dim3(nW  / 2048), blk, 0, stream>>>(Wk, wkb, nW);
    cvt_bf16_kernel<<<dim3(nW  / 2048), blk, 0, stream>>>(Wv, wvb, nW);
    cvt_bf16_kernel<<<dim3(nW  / 2048), blk, 0, stream>>>(Wo, wob, nW);

    // q = (x @ Wq^T + bq) * scale
    gemm_bt<true><<<dim3(Mq / 64, D / 64), blk, 0, stream>>>(
        xb, wqb, bq, scale, qv, Mq, D, D);
    // k = (xa @ Wk^T) * scale
    gemm_bt<true><<<dim3((Mk + 63) / 64, D / 64), blk, 0, stream>>>(
        xab, wkb, nullptr, scale, kv, Mk, D, D);
    // v = xa @ Wv^T + bv
    gemm_bt<true><<<dim3((Mk + 63) / 64, D / 64), blk, 0, stream>>>(
        xab, wvb, bv, 1.0f, vv, Mk, D, D);
    // attention: 8 waves x 32 queries = 256 queries per block
    attn_kernel<<<dim3(NB * 16 * (T / 256)), blk, 0, stream>>>(
        qv, kv, vv, av, T, T1, D, HD);
    // out = attn @ Wo^T + bo
    gemm_bt<false><<<dim3(Mq / 64, D / 64), blk, 0, stream>>>(
        av, wob, bo, 1.0f, (float*)d_out, Mq, D, D);
}
